// AttentionBlock_47794396070486
// MI455X (gfx1250) — compile-verified, matrix operand order checked by reading, not run on hardware
//
#include <hip/hip_runtime.h>
#include <stdint.h>

// ---------------------------------------------------------------------------
// AttentionBlock for MI455X (gfx1250): f16 WMMA flash-attention pipeline.
//   B=8, H=W=64 -> N=4096, C=256, d=32
// ---------------------------------------------------------------------------

typedef _Float16 h8  __attribute__((ext_vector_type(8)));
typedef _Float16 h16 __attribute__((ext_vector_type(16)));
typedef float    f4  __attribute__((ext_vector_type(4)));
typedef float    f8  __attribute__((ext_vector_type(8)));

constexpr int kB  = 8;
constexpr int kN  = 4096;   // H*W
constexpr int kC  = 256;
constexpr int kD  = 32;     // C/8
constexpr int kBM = 128;    // query rows per workgroup (8 waves x 16 rows)
constexpr int kBN = 64;     // keys per inner iteration (4 S-tiles, 32 PV WMMAs)
constexpr int kKP = 40;     // K LDS row stride (halves): 80B, conflict-free
constexpr int kVPD = 72;    // V/P LDS row stride (halves): 144B, conflict-free

static __device__ __forceinline__ f8 wmma16(h16 a, h16 b, f8 c) {
  return __builtin_amdgcn_wmma_f32_16x16x32_f16(false, a, false, b, (short)0, c,
                                                false, false);
}

static __device__ __forceinline__ h16 hcat(h8 lo, h8 hi) {
  h16 r;
#pragma unroll
  for (int i = 0; i < 8; ++i) { r[i] = lo[i]; r[i + 8] = hi[i]; }
  return r;
}

static __device__ __forceinline__ f8 fzero8() {
  f8 z;
#pragma unroll
  for (int i = 0; i < 8; ++i) z[i] = 0.0f;
  return z;
}

static __device__ __forceinline__ float fast_exp2(float x) {
#if __has_builtin(__builtin_amdgcn_exp2f)
  return __builtin_amdgcn_exp2f(x);   // v_exp_f32 (native exp2)
#else
  return exp2f(x);
#endif
}

// Load a 16x32 f16 WMMA operand whose reduction dim is contiguous in memory
// with row stride `stride` (halves). Matches the ISA 16-bit A/B layout:
//   lane l (l<16):  row l,    K = 0..7  then 16..23
//   lane l (l>=16): row l-16, K = 8..15 then 24..31
// -> two 16B loads per lane (global_load_b128 / ds_load_b128).
static __device__ __forceinline__ h16 ld_op(const _Float16* base, int stride,
                                            int lane) {
  const _Float16* p = base + (size_t)(lane & 15) * stride + (lane >> 4) * 8;
  return hcat(*(const h8*)p, *(const h8*)(p + 16));
}

// --------------------- async global->LDS staging (CDNA5) -------------------
#if __has_builtin(__builtin_amdgcn_global_load_async_to_lds_b128)
#define HAVE_ASYNC_LDS 1
typedef int v4i __attribute__((__vector_size__(16)));
typedef __attribute__((address_space(1))) v4i gv4i;   // global int4
typedef __attribute__((address_space(3))) v4i lv4i;   // LDS int4
static __device__ __forceinline__ void async_b128(const _Float16* g,
                                                  _Float16* l) {
  // Generic LDS pointer low 32 bits == LDS offset (ISA 10.2 aperture rules).
  unsigned loff = (unsigned)(uintptr_t)l;
  __builtin_amdgcn_global_load_async_to_lds_b128(
      (gv4i*)(uintptr_t)g, (lv4i*)loff, 0, 0);
}
static __device__ __forceinline__ void wait_async() {
#if __has_builtin(__builtin_amdgcn_s_wait_asynccnt)
  __builtin_amdgcn_s_wait_asynccnt(0);
#else
  asm volatile("s_wait_asynccnt 0x0" ::: "memory");
#endif
}
#else
#define HAVE_ASYNC_LDS 0
static __device__ __forceinline__ void async_b128(const _Float16* g,
                                                  _Float16* l) {
  *(h8*)l = *(const h8*)g;  // fallback: VGPR bounce
}
static __device__ __forceinline__ void wait_async() {}
#endif

// ---------------------------------------------------------------------------
// Kernel 0: transpose weights to f16 so WMMA B-operands are contiguous in K.
// ---------------------------------------------------------------------------
__global__ void prep_weights(const float* __restrict__ wq,
                             const float* __restrict__ wk,
                             const float* __restrict__ wv,
                             const float* __restrict__ wo,
                             _Float16* __restrict__ wqT, _Float16* __restrict__ wkT,
                             _Float16* __restrict__ wvT, _Float16* __restrict__ woT) {
  int t = blockIdx.x * blockDim.x + threadIdx.x;
  if (t >= kC * kC) return;
  int r = t >> 8, c = t & 255;
  wvT[t] = (_Float16)wv[c * kC + r];
  woT[t] = (_Float16)wo[c * kC + r];
  if (r < kD) {
    wqT[r * kC + c] = (_Float16)wq[c * kD + r];
    wkT[r * kC + c] = (_Float16)wk[c * kD + r];
  }
}

// ---------------------------------------------------------------------------
// Kernel 1: q/k/v projections. 256 WGs x 256 thr; each wave owns 16 rows.
// Outputs: qh/kh row-major [B*N][32] f16, vT per-batch-transposed [B][C][N] f16.
// ---------------------------------------------------------------------------
__global__ __launch_bounds__(256) void proj_qkv(
    const float* __restrict__ x, const _Float16* __restrict__ wqT,
    const _Float16* __restrict__ wkT, const _Float16* __restrict__ wvT,
    _Float16* __restrict__ qh, _Float16* __restrict__ kh,
    _Float16* __restrict__ vT) {
  const int wave = threadIdx.x >> 5, lane = threadIdx.x & 31;
  const int lr = lane & 15, hs = lane >> 4;
  const int rg = blockIdx.x * kBM + wave * 16;   // global row over B*N

  f8 aq[2], ak[2], av[16];
#pragma unroll
  for (int t = 0; t < 2; ++t) { aq[t] = fzero8(); ak[t] = fzero8(); }
#pragma unroll
  for (int t = 0; t < 16; ++t) av[t] = fzero8();

#pragma unroll 1
  for (int kc = 0; kc < kC / kD; ++kc) {
    const float* xp = x + (size_t)(rg + lr) * kC + kc * kD + hs * 8;
    f4 x0 = *(const f4*)(xp);      f4 x1 = *(const f4*)(xp + 4);
    f4 x2 = *(const f4*)(xp + 16); f4 x3 = *(const f4*)(xp + 20);
    h16 xa;
#pragma unroll
    for (int i = 0; i < 4; ++i) {
      xa[i]      = (_Float16)x0[i];
      xa[i + 4]  = (_Float16)x1[i];
      xa[i + 8]  = (_Float16)x2[i];
      xa[i + 12] = (_Float16)x3[i];
    }
#pragma unroll
    for (int t = 0; t < 2; ++t)
      aq[t] = wmma16(xa, ld_op(wqT + (size_t)t * 16 * kC + kc * kD, kC, lane), aq[t]);
#pragma unroll
    for (int t = 0; t < 2; ++t)
      ak[t] = wmma16(xa, ld_op(wkT + (size_t)t * 16 * kC + kc * kD, kC, lane), ak[t]);
#pragma unroll
    for (int t = 0; t < 16; ++t)
      av[t] = wmma16(xa, ld_op(wvT + (size_t)t * 16 * kC + kc * kD, kC, lane), av[t]);
  }

#pragma unroll
  for (int t = 0; t < 2; ++t)
#pragma unroll
    for (int r = 0; r < 8; ++r) {
      int row = rg + (hs ? r + 8 : r);
      qh[(size_t)row * kD + t * 16 + lr] = (_Float16)aq[t][r];
      kh[(size_t)row * kD + t * 16 + lr] = (_Float16)ak[t][r];
    }
#pragma unroll
  for (int t = 0; t < 16; ++t)
#pragma unroll
    for (int r = 0; r < 8; ++r) {
      int row = rg + (hs ? r + 8 : r);
      int b = row >> 12, n = row & (kN - 1);
      vT[((size_t)b * kC + t * 16 + lr) * kN + n] = (_Float16)av[t][r];
    }
}

// ---------------------------------------------------------------------------
// Kernel 2: fused flash attention, exp2-domain online softmax, async-LDS
// double buffering. One WG per (batch, 128-query block).
// ---------------------------------------------------------------------------
__global__ __launch_bounds__(256) void flash_attn(
    const _Float16* __restrict__ qh, const _Float16* __restrict__ kh,
    const _Float16* __restrict__ vT, _Float16* __restrict__ att) {
  __shared__ __align__(16) _Float16 Kb[2][kBN][kKP];   // 2 x 64 keys x d    (10 KB)
  __shared__ __align__(16) _Float16 Vb[2][kC][kVPD];   // 2 x V^T 256x64     (72 KB)
  __shared__ __align__(16) _Float16 Pb[8][16][kVPD];   // per-wave P scratch (18 KB)

  const int wave = threadIdx.x >> 5, lane = threadIdx.x & 31;
  const int tid = threadIdx.x;
  const int lr = lane & 15, hs = lane >> 4;
  const int b = blockIdx.x >> 5, qb = blockIdx.x & 31;
  const int qrow0 = qb * kBM + wave * 16;              // within batch

  const _Float16* kbase = kh + (size_t)b * kN * kD;
  const _Float16* vbase = vT + (size_t)b * kC * kN;

  // Q operand, pre-scaled by log2(e) so softmax runs in the exp2 domain.
  h16 qa = ld_op(qh + ((size_t)b * kN + qrow0) * kD, kD, lane);
#pragma unroll
  for (int i = 0; i < 16; ++i) qa[i] *= (_Float16)1.44269504088896340736f;

  f8 acc[16];
#pragma unroll
  for (int t = 0; t < 16; ++t) acc[t] = fzero8();
  float mi[8], li[8];
#pragma unroll
  for (int r = 0; r < 8; ++r) { mi[r] = -1e30f; li[r] = 0.0f; }

  const int krow = tid >> 2, kch = tid & 3;            // K staging: 16B/thread

  // Prologue: stage block 0 into buffer 0.
  async_b128(kbase + (size_t)krow * kD + kch * 8, &Kb[0][krow][kch * 8]);
  {
    const _Float16* vp = vbase + (size_t)tid * kN;
#pragma unroll
    for (int ch = 0; ch < 8; ++ch)
      async_b128(vp + ch * 8, &Vb[0][tid][ch * 8]);
  }

  const int nIter = kN / kBN;  // 64
  for (int kb = 0; kb < nIter; ++kb) {
    const int buf = kb & 1;
    wait_async();      // this thread's fills of buf are complete
    __syncthreads();   // all fills done; all reads of buf^1 (iter kb-1) done

    if (kb + 1 < nIter) {  // stage next block into the other buffer
      const int nb = buf ^ 1;
      async_b128(kbase + (size_t)((kb + 1) * kBN + krow) * kD + kch * 8,
                 &Kb[nb][krow][kch * 8]);
      const _Float16* vp = vbase + (size_t)tid * kN + (kb + 1) * kBN;
#pragma unroll
      for (int ch = 0; ch < 8; ++ch)
        async_b128(vp + ch * 8, &Vb[nb][tid][ch * 8]);
    }

    // --- S = Q K^T for 64 keys (four 16x16 tiles), log2 domain ---
    f8 s[4];
#pragma unroll
    for (int j = 0; j < 4; ++j)
      s[j] = wmma16(qa, ld_op(&Kb[buf][j * 16][0], kKP, lane), fzero8());

    // --- online softmax (rows spread over 16 lanes; VGPR r <-> row r/r+8) ---
    float sc[8];
#pragma unroll
    for (int r = 0; r < 8; ++r) {
      float m = fmaxf(fmaxf(s[0][r], s[1][r]), fmaxf(s[2][r], s[3][r]));
#pragma unroll
      for (int o = 8; o >= 1; o >>= 1) m = fmaxf(m, __shfl_xor(m, o, 16));
      float nm = fmaxf(mi[r], m);
      float p0 = fast_exp2(s[0][r] - nm);
      float p1 = fast_exp2(s[1][r] - nm);
      float p2 = fast_exp2(s[2][r] - nm);
      float p3 = fast_exp2(s[3][r] - nm);
      float rs = (p0 + p1) + (p2 + p3);
#pragma unroll
      for (int o = 8; o >= 1; o >>= 1) rs += __shfl_xor(rs, o, 16);
      sc[r] = fast_exp2(mi[r] - nm);
      li[r] = li[r] * sc[r] + rs;
      mi[r] = nm;
      int row = hs ? r + 8 : r;                // C-layout -> [row][key] in LDS
      Pb[wave][row][lr]      = (_Float16)p0;
      Pb[wave][row][16 + lr] = (_Float16)p1;
      Pb[wave][row][32 + lr] = (_Float16)p2;
      Pb[wave][row][48 + lr] = (_Float16)p3;
    }
#pragma unroll
    for (int t = 0; t < 16; ++t)
#pragma unroll
      for (int r = 0; r < 8; ++r) acc[t][r] *= sc[r];

    // --- O += P V  (wave-private LDS bounce; same-wave LDS ops in-order) ---
    const h16 pa0 = ld_op(&Pb[wave][0][0], kVPD, lane);   // keys  0..31
    const h16 pa1 = ld_op(&Pb[wave][0][32], kVPD, lane);  // keys 32..63
#pragma unroll
    for (int ct = 0; ct < 16; ++ct) {
      acc[ct] = wmma16(pa0, ld_op(&Vb[buf][ct * 16][0], kVPD, lane), acc[ct]);
      acc[ct] = wmma16(pa1, ld_op(&Vb[buf][ct * 16][32], kVPD, lane), acc[ct]);
    }
  }

  float inv[8];
#pragma unroll
  for (int r = 0; r < 8; ++r) inv[r] = 1.0f / li[r];
#pragma unroll
  for (int t = 0; t < 16; ++t)
#pragma unroll
    for (int r = 0; r < 8; ++r) {
      int row = qrow0 + (hs ? r + 8 : r);
      att[((size_t)b * kN + row) * kC + t * 16 + lr] =
          (_Float16)(acc[t][r] * inv[r]);
    }
}

// ---------------------------------------------------------------------------
// Kernel 3: out = x + attended @ w_out  (f16 WMMA, f32 residual+store)
// ---------------------------------------------------------------------------
__global__ __launch_bounds__(256) void out_proj(
    const _Float16* __restrict__ att, const _Float16* __restrict__ woT,
    const float* __restrict__ xin, float* __restrict__ out) {
  const int wave = threadIdx.x >> 5, lane = threadIdx.x & 31;
  const int lr = lane & 15, hs = lane >> 4;
  const int rg = blockIdx.x * kBM + wave * 16;

  f8 acc[16];
#pragma unroll
  for (int t = 0; t < 16; ++t) acc[t] = fzero8();

#pragma unroll 1
  for (int kc = 0; kc < kC / kD; ++kc) {
    h16 aA = ld_op(att + (size_t)rg * kC + kc * kD, kC, lane);
#pragma unroll
    for (int t = 0; t < 16; ++t)
      acc[t] = wmma16(aA, ld_op(woT + (size_t)t * 16 * kC + kc * kD, kC, lane),
                      acc[t]);
  }

#pragma unroll
  for (int t = 0; t < 16; ++t)
#pragma unroll
    for (int r = 0; r < 8; ++r) {
      int row = rg + (hs ? r + 8 : r);
      size_t idx = (size_t)row * kC + t * 16 + lr;
      out[idx] = xin[idx] + acc[t][r];
    }
}

// ---------------------------------------------------------------------------
extern "C" void kernel_launch(void* const* d_in, const int* in_sizes, int n_in,
                              void* d_out, int out_size, void* d_ws, size_t ws_size,
                              hipStream_t stream) {
  (void)in_sizes; (void)n_in; (void)out_size; (void)ws_size;
  const float* x  = (const float*)d_in[0];
  const float* wq = (const float*)d_in[1];
  const float* wk = (const float*)d_in[2];
  const float* wv = (const float*)d_in[3];
  const float* wo = (const float*)d_in[4];
  float* out = (float*)d_out;

  char* ws = (char*)d_ws;
  size_t off = 0;
  auto carve = [&](size_t bytes) {
    char* p = ws + off;
    off += (bytes + 255) & ~(size_t)255;
    return p;
  };
  _Float16* wqT = (_Float16*)carve((size_t)kD * kC * 2);          //  16 KB
  _Float16* wkT = (_Float16*)carve((size_t)kD * kC * 2);          //  16 KB
  _Float16* wvT = (_Float16*)carve((size_t)kC * kC * 2);          // 128 KB
  _Float16* woT = (_Float16*)carve((size_t)kC * kC * 2);          // 128 KB
  _Float16* qh  = (_Float16*)carve((size_t)kB * kN * kD * 2);     //   2 MB
  _Float16* kh  = (_Float16*)carve((size_t)kB * kN * kD * 2);     //   2 MB
  _Float16* vT  = (_Float16*)carve((size_t)kB * kC * kN * 2);     //  16 MB
  _Float16* att = (_Float16*)carve((size_t)kB * kN * kC * 2);     //  16 MB

  const int nblk = kB * kN / kBM;  // 256 workgroups
  prep_weights<<<(kC * kC + 255) / 256, 256, 0, stream>>>(wq, wk, wv, wo,
                                                          wqT, wkT, wvT, woT);
  proj_qkv<<<nblk, 256, 0, stream>>>(x, wqT, wkT, wvT, qh, kh, vT);
  flash_attn<<<nblk, 256, 0, stream>>>(qh, kh, vT, att);
  out_proj<<<nblk, 256, 0, stream>>>(att, woT, x, out);
}
